// EdgeNet_46952582480249
// MI455X (gfx1250) — compile-verified, matrix operand order checked by reading, not run on hardware
//
#include <hip/hip_runtime.h>
#include <hip/hip_bf16.h>
#include <hip/hip_fp16.h>

typedef _Float16 h16;
typedef __attribute__((ext_vector_type(16))) _Float16 v16h;
typedef __attribute__((ext_vector_type(4)))  _Float16 h4;
typedef __attribute__((ext_vector_type(8)))  float    v8f;

#define DD 13     // input features
#define HH 32     // hidden
#define HD 45     // H + D
#define K1 96     // 2*(H+D)=90 padded to 3*32
#define C1 64     // conv layer-1 out
#define K2 64
#define C2 32
#define FSTRIDE 48  // feat row stride (halves)

// ---------------------------------------------------------------- utilities
__global__ void zero_f32(float* __restrict__ p, int n) {
  int i = blockIdx.x * 256 + threadIdx.x;
  if (i < n) p[i] = 0.f;
}

// ------------------------------------------------------------- batchnorm stats
__global__ void bn_stats(const float* __restrict__ x, int N,
                         float* __restrict__ mu, float* __restrict__ istd) {
  __shared__ float ss[256], sq[256];
  int d = blockIdx.x, t = threadIdx.x;
  float s = 0.f, q = 0.f;
  for (int r = t; r < N; r += 256) {
    float v = x[(size_t)r * DD + d];
    s += v; q += v * v;
  }
  ss[t] = s; sq[t] = q;
  __syncthreads();
  for (int o = 128; o > 0; o >>= 1) {
    if (t < o) { ss[t] += ss[t + o]; sq[t] += sq[t + o]; }
    __syncthreads();
  }
  if (t == 0) {
    float m = ss[0] / (float)N;
    float v = sq[0] / (float)N - m * m;
    mu[d] = m;
    istd[d] = rsqrtf(v + 1e-5f);
  }
}

// ------------------------------------- pad + transpose conv weights to fp16
// wt1[col][k] : col<64, k<96 (k>=90 zero);  wt2[col][k] : col<32, k<64
__global__ void prep_w(const float* __restrict__ w1, const float* __restrict__ w2,
                       h16* __restrict__ wt1, h16* __restrict__ wt2) {
  int i = blockIdx.x * 256 + threadIdx.x;
  if (i < C1 * K1) {
    int col = i / K1, k = i % K1;
    wt1[i] = (k < 2 * HD) ? (h16)w1[k * C1 + col] : (h16)0.f;
  }
  int j = i - C1 * K1;
  if (j >= 0 && j < C2 * K2) {
    int col = j / K2, k = j % K2;
    wt2[j] = (h16)w2[k * C2 + col];
  }
}

// --------------------------------------------- BN + inputnet + feat assembly
__global__ __launch_bounds__(128) void input_net(
    const float* __restrict__ x, const float* __restrict__ mu,
    const float* __restrict__ istd, const float* __restrict__ gamma,
    const float* __restrict__ beta,
    const float* __restrict__ w1, const float* __restrict__ b1,
    const float* __restrict__ w2, const float* __restrict__ b2,
    h16* __restrict__ feat, int N) {
  __shared__ float xn_s[4][16];
  __shared__ float h1_s[4][32];
  int tid = threadIdx.x, wave = tid >> 5, j = tid & 31;
  int node = blockIdx.x * 4 + wave;
  bool valid = node < N;
  int nn = valid ? node : 0;
  if (j < DD) {
    float v = (x[(size_t)nn * DD + j] - mu[j]) * istd[j] * gamma[j] + beta[j];
    xn_s[wave][j] = v;
  }
  __syncthreads();
  float s = b1[j];
  #pragma unroll
  for (int k = 0; k < DD; ++k) s += xn_s[wave][k] * w1[k * HH + j];
  h1_s[wave][j] = fmaxf(s, 0.f);
  __syncthreads();
  float s2 = b2[j];
  #pragma unroll
  for (int k = 0; k < HH; ++k) s2 += h1_s[wave][k] * w2[k * HH + j];
  float h2 = tanhf(s2);
  if (valid) {
    size_t rb = (size_t)node * FSTRIDE;
    feat[rb + j] = (h16)h2;                                  // cols 0..31 = Hn
    if (j < DD)      feat[rb + HH + j] = (h16)xn_s[wave][j]; // cols 32..44 = Xn
    else if (j < 16) feat[rb + HH + j] = (h16)0.f;           // cols 45..47 = 0
  }
}

// --------------------------------------------------- EdgeConv MLP via WMMA
__global__ __launch_bounds__(128) void edge_mlp(
    const int* __restrict__ ei, const h16* __restrict__ feat,
    const h16* __restrict__ wt1, const h16* __restrict__ wt2,
    const float* __restrict__ cb1, const float* __restrict__ cb2,
    float* __restrict__ Hn, int E) {
  __shared__ h16 sW1[C1 * K1];     // [col][k], 12 KB
  __shared__ h16 sW2[C2 * K2];     // [col][k],  4 KB
  __shared__ float sB1[C1];
  __shared__ float sB2[C2];
  __shared__ h16 sM[4][16 * K1];   // per-wave A tile (reused as h-tile)
  __shared__ int sDst[4][16];

  int tid = threadIdx.x;
  for (int i = tid; i < C1 * K1; i += 128) sW1[i] = wt1[i];
  for (int i = tid; i < C2 * K2; i += 128) sW2[i] = wt2[i];
  if (tid < C1) sB1[tid] = cb1[tid];
  if (tid < C2) sB2[tid] = cb2[tid];

  int wave = tid >> 5, lane = tid & 31;
  int ebase = (blockIdx.x * 4 + wave) * 16;

  // ---- gather: 2 lanes per edge, 8B vector loads from L2-resident feat
  int el = lane >> 1, hp = lane & 1;
  int e = ebase + el;
  bool valid = (e < E);
  int ec = valid ? e : 0;
  int src = ei[ec], dst = ei[E + ec];
  if (hp == 0) sDst[wave][el] = valid ? dst : -1;
  h16* me = &sM[wave][el * K1];
  const h16* fi = feat + (size_t)dst * FSTRIDE;   // x_i (target)
  const h16* fj = feat + (size_t)src * FSTRIDE;   // x_j (source)
  for (int c = hp; c < 12; c += 2) {
    h4 va = *(const h4*)(fi + 4 * c);
    h4 vb = *(const h4*)(fj + 4 * c);
    #pragma unroll
    for (int t = 0; t < 4; ++t) {
      int f = 4 * c + t;
      if (f < HD) { me[f] = va[t]; me[HD + f] = (h16)(vb[t] - va[t]); }
    }
  }
  if (hp == 0) {
    #pragma unroll
    for (int p = 2 * HD; p < K1; ++p) me[p] = (h16)0.f;   // K pad 90..95
  }
  __syncthreads();

  // ---- layer 1: [16,96] x [96,64] in fp16, f32 accumulate
  int lh = lane >> 4;   // half of wave: K-block selector
  int cl = lane & 15;   // row for A, col for B/C
  const h16* arow = &sM[wave][cl * K1];

  v8f acc[4] = {};
  #pragma unroll
  for (int s = 0; s < 3; ++s) {
    v16h a;
    const h16* ap = arow + s * 32 + 8 * lh;
    #pragma unroll
    for (int v = 0; v < 8; ++v) {
      int kk = (v < 4) ? 2 * v : 8 + 2 * v;
      a[2 * v] = ap[kk]; a[2 * v + 1] = ap[kk + 1];
    }
    #pragma unroll
    for (int n = 0; n < 4; ++n) {
      v16h b;
      const h16* bp = &sW1[(n * 16 + cl) * K1 + s * 32 + 8 * lh];
      #pragma unroll
      for (int v = 0; v < 8; ++v) {
        int kk = (v < 4) ? 2 * v : 8 + 2 * v;
        b[2 * v] = bp[kk]; b[2 * v + 1] = bp[kk + 1];
      }
      acc[n] = __builtin_amdgcn_wmma_f32_16x16x32_f16(
          false, a, false, b, (short)0, acc[n], false, false);
    }
  }

  __syncthreads();
  // ---- bias + relu, restage as fp16 A tile [16][64]
  h16* hbuf = &sM[wave][0];
  #pragma unroll
  for (int n = 0; n < 4; ++n)
    #pragma unroll
    for (int r = 0; r < 8; ++r) {
      float t = acc[n][r] + sB1[n * 16 + cl];
      hbuf[(r + 8 * lh) * C1 + n * 16 + cl] = (h16)fmaxf(t, 0.f);
    }
  __syncthreads();

  // ---- layer 2: [16,64] x [64,32]
  v8f o[2] = {};
  const h16* arow2 = &hbuf[cl * C1];
  #pragma unroll
  for (int s = 0; s < 2; ++s) {
    v16h a;
    const h16* ap = arow2 + s * 32 + 8 * lh;
    #pragma unroll
    for (int v = 0; v < 8; ++v) {
      int kk = (v < 4) ? 2 * v : 8 + 2 * v;
      a[2 * v] = ap[kk]; a[2 * v + 1] = ap[kk + 1];
    }
    #pragma unroll
    for (int n = 0; n < 2; ++n) {
      v16h b;
      const h16* bp = &sW2[(n * 16 + cl) * K2 + s * 32 + 8 * lh];
      #pragma unroll
      for (int v = 0; v < 8; ++v) {
        int kk = (v < 4) ? 2 * v : 8 + 2 * v;
        b[2 * v] = bp[kk]; b[2 * v + 1] = bp[kk + 1];
      }
      o[n] = __builtin_amdgcn_wmma_f32_16x16x32_f16(
          false, a, false, b, (short)0, o[n], false, false);
    }
  }

  // ---- bias + tanh + scatter-add (segment_sum over dst)
  #pragma unroll
  for (int n = 0; n < 2; ++n)
    #pragma unroll
    for (int r = 0; r < 8; ++r) {
      int m = r + 8 * lh;
      int d = sDst[wave][m];
      if (d >= 0) {
        float v = tanhf(o[n][r] + sB2[n * 16 + cl]);
        atomicAdd(&Hn[(size_t)d * HH + n * 16 + cl], v);
      }
    }
}

// ------------------------------------------------- global mean pool (accum)
// pool rows stride 48: [0..44]=sums, [45]=count
__global__ void pool_acc(const float* __restrict__ Hn, const h16* __restrict__ feat,
                         const int* __restrict__ batch, float* __restrict__ pool,
                         int N) {
  int i = blockIdx.x * 256 + threadIdx.x;
  if (i >= N * 64) return;
  int n = i >> 6, f = i & 63;
  int g = batch[n];
  if (f < HH)      atomicAdd(&pool[g * 48 + f], Hn[(size_t)n * HH + f]);
  else if (f < HD) atomicAdd(&pool[g * 48 + f], (float)feat[(size_t)n * FSTRIDE + f]);
  else if (f == HD) atomicAdd(&pool[g * 48 + 45], 1.0f);
}

// ----------------------------------------------------- outputnet + sigmoid
__global__ void out_net(const float* __restrict__ pool,
                        const float* __restrict__ w1, const float* __restrict__ b1,
                        const float* __restrict__ w2, const float* __restrict__ b2,
                        float* __restrict__ out, int G) {
  int g = blockIdx.x * blockDim.x + threadIdx.x;
  if (g >= G) return;
  float c = pool[g * 48 + 45];
  c = (c < 1.f) ? 1.f : c;
  float xm[HD];
  #pragma unroll
  for (int f = 0; f < HD; ++f) xm[f] = pool[g * 48 + f] / c;
  float o = b2[0];
  #pragma unroll
  for (int j = 0; j < HH; ++j) {
    float s = b1[j];
    #pragma unroll
    for (int f = 0; f < HD; ++f) s += xm[f] * w1[f * HH + j];
    o += fmaxf(s, 0.f) * w2[j];
  }
  out[g] = 1.f / (1.f + expf(-o));
}

// --------------------------------------------------------------------------
extern "C" void kernel_launch(void* const* d_in, const int* in_sizes, int n_in,
                              void* d_out, int out_size, void* d_ws, size_t ws_size,
                              hipStream_t stream) {
  const float* x       = (const float*)d_in[0];
  const int*   ei      = (const int*)  d_in[1];
  const int*   batch   = (const int*)  d_in[2];
  const float* gamma   = (const float*)d_in[3];
  const float* beta    = (const float*)d_in[4];
  const float* in_w1   = (const float*)d_in[5];
  const float* in_b1   = (const float*)d_in[6];
  const float* in_w2   = (const float*)d_in[7];
  const float* in_b2   = (const float*)d_in[8];
  const float* conv_w1 = (const float*)d_in[9];
  const float* conv_b1 = (const float*)d_in[10];
  const float* conv_w2 = (const float*)d_in[11];
  const float* conv_b2 = (const float*)d_in[12];
  const float* out_w1  = (const float*)d_in[13];
  const float* out_b1  = (const float*)d_in[14];
  const float* out_w2  = (const float*)d_in[15];
  const float* out_b2  = (const float*)d_in[16];
  (void)n_in; (void)ws_size;

  int N = in_sizes[0] / DD;
  int E = in_sizes[1] / 2;
  int G = out_size;

  char* base = (char*)d_ws;
  size_t off = 0;
  auto walloc = [&](size_t bytes) -> void* {
    void* p = base + off;
    off = (off + bytes + 255) & ~(size_t)255;
    return p;
  };
  float* mu   = (float*)walloc(DD * sizeof(float));
  float* istd = (float*)walloc(DD * sizeof(float));
  float* Hn   = (float*)walloc((size_t)N * HH * sizeof(float));
  float* pool = (float*)walloc((size_t)G * 48 * sizeof(float));
  h16*   feat = (h16*)  walloc((size_t)N * FSTRIDE * sizeof(h16));
  h16*   wt1  = (h16*)  walloc((size_t)C1 * K1 * sizeof(h16));
  h16*   wt2  = (h16*)  walloc((size_t)C2 * K2 * sizeof(h16));

  zero_f32<<<(N * HH + 255) / 256, 256, 0, stream>>>(Hn, N * HH);
  zero_f32<<<(G * 48 + 255) / 256, 256, 0, stream>>>(pool, G * 48);
  bn_stats<<<DD, 256, 0, stream>>>(x, N, mu, istd);
  prep_w<<<(C1 * K1 + C2 * K2 + 255) / 256, 256, 0, stream>>>(conv_w1, conv_w2, wt1, wt2);
  input_net<<<(N + 3) / 4, 128, 0, stream>>>(x, mu, istd, gamma, beta,
                                             in_w1, in_b1, in_w2, in_b2, feat, N);
  int tiles = (E + 15) / 16;
  edge_mlp<<<(tiles + 3) / 4, 128, 0, stream>>>(ei, feat, wt1, wt2,
                                                conv_b1, conv_b2, Hn, E);
  pool_acc<<<(N * 64 + 255) / 256, 256, 0, stream>>>(Hn, feat, batch, pool, N);
  out_net<<<(G + 127) / 128, 128, 0, stream>>>(pool, out_w1, out_b1, out_w2, out_b2,
                                               (float*)d_out, G);
}
